// MultiHeadAttention_50062138802817
// MI455X (gfx1250) — matrix operand order repaired and checked by reading, not compiled, not run on hardware
//
#include <hip/hip_runtime.h>
#include <hip/hip_bf16.h>
#include <math.h>

// ---------------------------------------------------------------------------
// MultiHeadAttention forward for MI455X (gfx1250, wave32, WMMA).
// B=1, T=4096, D=768, H=12, HD=64.
// Outputs (concatenated in d_out): output (T*D f32), weights (H*T*T f32).
//
// Roofline: weights tensor (805 MB) dominates -> minimize touches:
//   scores: write lower triangle only; softmax: read [0,m], write zeros above;
//   PV: read lower triangle once (A-fragment reused across all 4 N-tiles).
// ---------------------------------------------------------------------------

constexpr int Tn  = 4096;
constexpr int Dn  = 768;
constexpr int Hn  = 12;
constexpr int HDn = 64;

// Toggle for the async-to-LDS staging path in the projection GEMM.
#define USE_ASYNC_LDS 1

typedef __bf16 bf16;
typedef __attribute__((ext_vector_type(16))) __bf16 v16bf;
typedef __attribute__((ext_vector_type(8)))  float  v8f;

// ---------------------------------------------------------------------------
// WMMA fragment helpers (layouts per CDNA5 ISA 7.12.2, wave32).
// 16-bit A (16x32): lane holds row M = lane%16; half = lane/16:
//   two contiguous 8-element runs: K = half*8+{0..7} and K = 16+half*8+{0..7}.
// 16-bit B (32x16): lane supplies column N = lane%16; same two 8-element runs as A:
//   K = half*8+{0..7} and K = 16+half*8+{0..7}.
// f32 C/D (16x16): lane owns column N = lane%16; VGPR r -> row (lane/16)*8+r.
// ---------------------------------------------------------------------------

__device__ __forceinline__ v16bf frag_a_bf16(const bf16* __restrict__ A, int lda, int lane) {
  const int half = lane >> 4;
  const int m    = lane & 15;
  const bf16* row = A + (size_t)m * lda;
  v16bf f;
#pragma unroll
  for (int e = 0; e < 8; ++e) {
    f[e]     = row[half * 8 + e];
    f[8 + e] = row[16 + half * 8 + e];
  }
  return f;
}

__device__ __forceinline__ v16bf frag_a_f32(const float* __restrict__ A, int lda, int lane) {
  const int half = lane >> 4;
  const int m    = lane & 15;
  const float* row = A + (size_t)m * lda;
  v16bf f;
#pragma unroll
  for (int e = 0; e < 8; ++e) {
    f[e]     = (bf16)row[half * 8 + e];
    f[8 + e] = (bf16)row[16 + half * 8 + e];
  }
  return f;
}

// BT stored (N x K) row-major: row n of BT supplies column n of B.
__device__ __forceinline__ v16bf frag_b_bf16(const bf16* __restrict__ BT, int ldb, int lane) {
  const int half = lane >> 4;
  const int n    = lane & 15;
  const bf16* row = BT + (size_t)n * ldb + half * 8;
  v16bf f;
#pragma unroll
  for (int e = 0; e < 8; ++e) {
    f[e]     = row[e];
    f[8 + e] = row[16 + e];
  }
  return f;
}

__device__ __forceinline__ v8f wmma_bf16(v16bf a, v16bf b, v8f c) {
  return __builtin_amdgcn_wmma_f32_16x16x32_bf16(false, a, false, b, (short)0, c, false, false);
}

// ---------------------------------------------------------------------------
// Kernel 0: f32 -> bf16 conversion (grid-stride).
// ---------------------------------------------------------------------------
__global__ void __launch_bounds__(256)
cvt_f32_bf16(const float* __restrict__ src, bf16* __restrict__ dst, size_t n) {
  const size_t stride = (size_t)gridDim.x * blockDim.x;
  for (size_t i = (size_t)blockIdx.x * blockDim.x + threadIdx.x; i < n; i += stride)
    dst[i] = (bf16)src[i];
}

// ---------------------------------------------------------------------------
// Kernel 1: C[M,N] = scale * (A[M,K] @ W[N,K]^T + bias[N])   (torch Linear)
// Block: 256 threads = 8 waves as 2(m) x 4(n); block tile 64 x 128,
// wave tile 32 x 32 (2x2 register blocking, 4 WMMAs per K-step).
// The 64x32 A-block is staged into LDS once per K-step with
// global_load_async_to_lds_b128 and shared by the 4 n-waves.
// ---------------------------------------------------------------------------
template <bool OUT_BF16>
__global__ void __launch_bounds__(256)
gemm_xwT(const bf16* __restrict__ A, const bf16* __restrict__ W,
         const float* __restrict__ bias, void* __restrict__ out,
         int M, int N, int K, float scale) {
  __shared__ bf16 As[64][32];

  const int lane = threadIdx.x & 31;
  const int wave = threadIdx.x >> 5;
  const int wm = wave & 1;         // 2 m-groups
  const int wn = wave >> 1;        // 4 n-groups
  const int mblock = blockIdx.y * 64;
  const int n0 = blockIdx.x * 128 + wn * 32;

  // A-staging assignment: thread t copies 16B (8 bf16) of a 64x32 tile.
  const int t    = threadIdx.x;
  const int arow = t >> 2;                 // 0..63
  const int acol = (t & 3) * 8;            // 0,8,16,24
  const bf16* gA = A + (size_t)(mblock + arow) * K + acol;
#if USE_ASYNC_LDS
  const unsigned lds_addr = (unsigned)(uintptr_t)&As[arow][acol];
#endif

  const bf16* W0r = W + (size_t)n0 * K;
  const bf16* W1r = W + (size_t)(n0 + 16) * K;

  v8f acc00 = {}, acc01 = {}, acc10 = {}, acc11 = {};

  for (int k0 = 0; k0 < K; k0 += 32) {
#if USE_ASYNC_LDS
    // Async DMA: mem -> LDS, tracked by ASYNCcnt (ISA 08_async_tensor §4).
    asm volatile("global_load_async_to_lds_b128 %0, %1, off"
                 :: "v"(lds_addr), "v"(gA + k0) : "memory");
    asm volatile("s_wait_asynccnt 0x0" ::: "memory");
#else
    *(reinterpret_cast<__attribute__((ext_vector_type(4))) unsigned*>(&As[arow][acol])) =
        *(const __attribute__((ext_vector_type(4))) unsigned*)(gA + k0);
#endif
    __syncthreads();

    __builtin_prefetch(W0r + k0 + 512, 0, 0);  // global_prefetch_b8

    // A fragments from LDS (ds_load), shared across the 4 n-waves.
    const int half = lane >> 4;
    const int ml   = lane & 15;
    v16bf a0, a1;
    {
      const bf16* r0 = &As[wm * 32 + ml][0];
      const bf16* r1 = &As[wm * 32 + 16 + ml][0];
#pragma unroll
      for (int e = 0; e < 8; ++e) {
        a0[e]     = r0[half * 8 + e];
        a0[8 + e] = r0[16 + half * 8 + e];
        a1[e]     = r1[half * 8 + e];
        a1[8 + e] = r1[16 + half * 8 + e];
      }
    }
    v16bf b0 = frag_b_bf16(W0r + k0, K, lane);
    v16bf b1 = frag_b_bf16(W1r + k0, K, lane);

    acc00 = wmma_bf16(a0, b0, acc00);
    acc01 = wmma_bf16(a0, b1, acc01);
    acc10 = wmma_bf16(a1, b0, acc10);
    acc11 = wmma_bf16(a1, b1, acc11);

    __syncthreads();  // protect As before next stage
  }

  const int n  = lane & 15;
  const int mb = (lane >> 4) * 8;
  const int m0 = mblock + wm * 32;
#pragma unroll
  for (int mi = 0; mi < 2; ++mi) {
#pragma unroll
    for (int nj = 0; nj < 2; ++nj) {
      const v8f& acc = mi == 0 ? (nj == 0 ? acc00 : acc01) : (nj == 0 ? acc10 : acc11);
      const int nn = n0 + nj * 16 + n;
      const float bv = bias[nn];
#pragma unroll
      for (int r = 0; r < 8; ++r) {
        const float val = (acc[r] + bv) * scale;
        const size_t idx = (size_t)(m0 + mi * 16 + mb + r) * N + nn;
        if (OUT_BF16) ((bf16*)out)[idx] = (bf16)val;
        else          ((float*)out)[idx] = val;
      }
    }
  }
}

// ---------------------------------------------------------------------------
// Kernel 2: causal scores, lower triangle only.  S[h] = Qh @ Kh^T
// (1/sqrt(HD) folded into Q).  One 32x32 tile per wave, triangular-decoded
// from a linear tile index; 8 WMMAs per wave.  All branches wave-uniform.
// ---------------------------------------------------------------------------
constexpr int NT32  = Tn / 32;                  // 128
constexpr int TRI32 = NT32 * (NT32 + 1) / 2;    // 8256

__device__ __forceinline__ void store_scores(float* __restrict__ out, const v8f& acc,
                                             int mm0, int nn0, int lane) {
  const int n     = nn0 + (lane & 15);
  const int mbase = mm0 + (lane >> 4) * 8;
#pragma unroll
  for (int r = 0; r < 8; ++r) {
    const int mm = mbase + r;
    out[(size_t)mm * Tn + n] = (n > mm) ? -__builtin_inff() : acc[r];
  }
}

__global__ void __launch_bounds__(256)
scores_kernel(const bf16* __restrict__ q, const bf16* __restrict__ k,
              float* __restrict__ wts) {
  const int lane = threadIdx.x & 31;
  const int wave = threadIdx.x >> 5;
  int tile = blockIdx.x * 8 + wave;           // [0, Hn*TRI32)
  const int h = tile / TRI32;
  const int t = tile - h * TRI32;

  // triangular decode: t -> (tm, tn), tn <= tm
  int tm = (int)((sqrtf(8.0f * (float)t + 1.0f) - 1.0f) * 0.5f);
  while ((tm + 1) * (tm + 2) / 2 <= t) ++tm;
  while (tm * (tm + 1) / 2 > t) --tm;
  const int tn = t - tm * (tm + 1) / 2;

  const int m0 = tm * 32;
  const int n0 = tn * 32;

  const bf16* qr0 = q + (size_t)m0 * Dn + h * HDn;
  const bf16* qr1 = qr0 + (size_t)16 * Dn;
  const bf16* kr0 = k + (size_t)n0 * Dn + h * HDn;
  const bf16* kr1 = kr0 + (size_t)16 * Dn;

  v8f acc00 = {}, acc01 = {}, acc10 = {}, acc11 = {};
#pragma unroll
  for (int k0 = 0; k0 < HDn; k0 += 32) {
    v16bf a0 = frag_a_bf16(qr0 + k0, Dn, lane);
    v16bf a1 = frag_a_bf16(qr1 + k0, Dn, lane);
    v16bf b0 = frag_b_bf16(kr0 + k0, Dn, lane);
    v16bf b1 = frag_b_bf16(kr1 + k0, Dn, lane);
    acc00 = wmma_bf16(a0, b0, acc00);
    acc01 = wmma_bf16(a0, b1, acc01);
    acc10 = wmma_bf16(a1, b0, acc10);
    acc11 = wmma_bf16(a1, b1, acc11);
  }

  float* out = wts + (size_t)h * Tn * Tn;
  store_scores(out, acc00, m0,      n0,      lane);
  store_scores(out, acc01, m0,      n0 + 16, lane);
  store_scores(out, acc10, m0 + 16, n0,      lane);
  store_scores(out, acc11, m0 + 16, n0 + 16, lane);
}

// ---------------------------------------------------------------------------
// Kernel 3: row softmax.  Row m reads only the valid prefix [0, m] and writes
// zeros over the masked suffix (m, T) — never touches the -inf region.
// ---------------------------------------------------------------------------
__global__ void __launch_bounds__(256)
softmax_rows(float* __restrict__ wts) {
  const int row = blockIdx.x;         // h*T + m
  const int m   = row & (Tn - 1);
  const int len = m + 1;
  float* p = wts + (size_t)row * Tn;
  __shared__ float red[256];
  const int tid = threadIdx.x;

  float mx = -__builtin_inff();
  for (int i = tid; i < len; i += 256) mx = fmaxf(mx, p[i]);
  red[tid] = mx;
  __syncthreads();
  for (int s = 128; s > 0; s >>= 1) {
    if (tid < s) red[tid] = fmaxf(red[tid], red[tid + s]);
    __syncthreads();
  }
  mx = red[0];
  __syncthreads();

  float sum = 0.f;
  for (int i = tid; i < len; i += 256) {
    const float e = __expf(p[i] - mx);
    p[i] = e;
    sum += e;
  }
  red[tid] = sum;
  __syncthreads();
  for (int s = 128; s > 0; s >>= 1) {
    if (tid < s) red[tid] += red[tid + s];
    __syncthreads();
  }
  const float inv = 1.0f / red[0];
  __syncthreads();

  for (int i = tid; i < len; i += 256) p[i] *= inv;
  for (int i = len + tid; i < Tn; i += 256) p[i] = 0.f;   // masked suffix
}

// ---------------------------------------------------------------------------
// Kernel 4: vT[h][n][t] = v[t, h*64+n]  (per-head transpose -> contiguous-K
// B-operand for PV).
// ---------------------------------------------------------------------------
__global__ void __launch_bounds__(256)
transpose_v(const bf16* __restrict__ v, bf16* __restrict__ vT) {
  const size_t total  = (size_t)Hn * HDn * Tn;
  const size_t stride = (size_t)gridDim.x * blockDim.x;
  for (size_t i = (size_t)blockIdx.x * blockDim.x + threadIdx.x; i < total; i += stride) {
    const int t = (int)(i % Tn);
    const int c = (int)(i / Tn);  // c = h*HD + n
    vT[i] = v[(size_t)t * Dn + c];
  }
}

// ---------------------------------------------------------------------------
// Kernel 5: attn[h] = P[h] (TxT f32 probs) @ Vh (Tx64).
// One wave computes the full 16 x 64 head-slice: the (expensive) f32 A
// fragment is loaded ONCE per K-step and reused across 4 B fragments
// (4x less weights traffic).  K-loop clipped to causal bound.
// ---------------------------------------------------------------------------
__global__ void __launch_bounds__(256)
attn_pv_kernel(const float* __restrict__ wts, const bf16* __restrict__ vT,
               bf16* __restrict__ attn) {
  const int lane = threadIdx.x & 31;
  const int wave = threadIdx.x >> 5;
  int tile = blockIdx.x * 8 + wave;        // [0, Hn * T/16)
  const int h  = tile / (Tn / 16);
  const int tm = tile - h * (Tn / 16);

  const float* wrow  = wts + (size_t)h * Tn * Tn + (size_t)(tm * 16) * Tn;
  const bf16*  vbase = vT + (size_t)h * HDn * Tn;

  int kend = (tm * 16 + 16 + 31) & ~31;    // causal bound, rounded up to 32
  if (kend > Tn) kend = Tn;

  v8f acc[4] = {{}, {}, {}, {}};
  for (int k0 = 0; k0 < kend; k0 += 32) {
    const v16bf a = frag_a_f32(wrow + k0, Tn, lane);
#pragma unroll
    for (int j = 0; j < 4; ++j)
      acc[j] = wmma_bf16(a, frag_b_bf16(vbase + (size_t)(j * 16) * Tn + k0, Tn, lane), acc[j]);
  }

  const int n  = lane & 15;
  const int mb = (lane >> 4) * 8;
#pragma unroll
  for (int j = 0; j < 4; ++j)
#pragma unroll
    for (int r = 0; r < 8; ++r)
      attn[(size_t)(tm * 16 + mb + r) * Dn + h * HDn + j * 16 + n] = (bf16)acc[j][r];
}

// ---------------------------------------------------------------------------
// Host-side orchestration.
// d_in order: query, key, value, mask, Wq_w, Wq_b, Wk_w, Wk_b, Wv_w, Wv_b,
//             W0_w, W0_b.   (mask unused: causal structure is analytic)
// ---------------------------------------------------------------------------
extern "C" void kernel_launch(void* const* d_in, const int* in_sizes, int n_in,
                              void* d_out, int out_size, void* d_ws, size_t ws_size,
                              hipStream_t stream) {
  (void)in_sizes; (void)n_in; (void)out_size; (void)ws_size;

  const float* x_q  = (const float*)d_in[0];
  const float* x_k  = (const float*)d_in[1];
  const float* x_v  = (const float*)d_in[2];
  const float* Wq_w = (const float*)d_in[4];
  const float* Wq_b = (const float*)d_in[5];
  const float* Wk_w = (const float*)d_in[6];
  const float* Wk_b = (const float*)d_in[7];
  const float* Wv_w = (const float*)d_in[8];
  const float* Wv_b = (const float*)d_in[9];
  const float* W0_w = (const float*)d_in[10];
  const float* W0_b = (const float*)d_in[11];

  float* outp = (float*)d_out;                 // (T, D) f32
  float* wts  = outp + (size_t)Tn * Dn;        // (H, T, T) f32

  const size_t td = (size_t)Tn * Dn;
  const size_t dd = (size_t)Dn * Dn;
  char* ws = (char*)d_ws;
  bf16* xq_b = (bf16*)ws;              ws += td * sizeof(bf16);
  bf16* xk_b = (bf16*)ws;              ws += td * sizeof(bf16);
  bf16* xv_b = (bf16*)ws;              ws += td * sizeof(bf16);
  bf16* wq_b = (bf16*)ws;              ws += dd * sizeof(bf16);
  bf16* wk_b = (bf16*)ws;              ws += dd * sizeof(bf16);
  bf16* wv_b = (bf16*)ws;              ws += dd * sizeof(bf16);
  bf16* w0_b = (bf16*)ws;              ws += dd * sizeof(bf16);
  bf16* q_b  = (bf16*)ws;              ws += td * sizeof(bf16);
  bf16* k_b  = (bf16*)ws;              ws += td * sizeof(bf16);
  bf16* v_b  = (bf16*)ws;              ws += td * sizeof(bf16);
  bf16* vT_b = (bf16*)ws;              ws += td * sizeof(bf16);
  bf16* at_b = (bf16*)ws;              ws += td * sizeof(bf16);

  const dim3 blk(256);

  // 0) conversions
  cvt_f32_bf16<<<4096, blk, 0, stream>>>(x_q,  xq_b, td);
  cvt_f32_bf16<<<4096, blk, 0, stream>>>(x_k,  xk_b, td);
  cvt_f32_bf16<<<4096, blk, 0, stream>>>(x_v,  xv_b, td);
  cvt_f32_bf16<<<2048, blk, 0, stream>>>(Wq_w, wq_b, dd);
  cvt_f32_bf16<<<2048, blk, 0, stream>>>(Wk_w, wk_b, dd);
  cvt_f32_bf16<<<2048, blk, 0, stream>>>(Wv_w, wv_b, dd);
  cvt_f32_bf16<<<2048, blk, 0, stream>>>(W0_w, w0_b, dd);

  // 1) projections (1/sqrt(HD) folded into Q).  Block tile 64x128.
  const dim3 gproj(Dn / 128, Tn / 64);
  gemm_xwT<true><<<gproj, blk, 0, stream>>>(xq_b, wq_b, Wq_b, q_b, Tn, Dn, Dn, 0.125f);
  gemm_xwT<true><<<gproj, blk, 0, stream>>>(xk_b, wk_b, Wk_b, k_b, Tn, Dn, Dn, 1.0f);
  gemm_xwT<true><<<gproj, blk, 0, stream>>>(xv_b, wv_b, Wv_b, v_b, Tn, Dn, Dn, 1.0f);

  // 2) causal scores, lower-triangle 32x32 tiles only
  scores_kernel<<<Hn * TRI32 / 8, blk, 0, stream>>>(q_b, k_b, wts);

  // 3) row softmax (prefix-only read, zero-fill masked suffix)
  softmax_rows<<<Hn * Tn, blk, 0, stream>>>(wts);

  // 4) V transpose, then P @ V with A-fragment reuse
  transpose_v<<<2048, blk, 0, stream>>>(v_b, vT_b);
  attn_pv_kernel<<<Hn * (Tn / 16) / 8, blk, 0, stream>>>(wts, vT_b, at_b);

  // 5) output projection (f32 out)
  gemm_xwT<false><<<gproj, blk, 0, stream>>>(at_b, w0_b, W0_b, outp, Tn, Dn, Dn, 1.0f);
}
